// Linear8bit_42640435315472
// MI455X (gfx1250) — matrix operand order repaired and checked by reading, not compiled, not run on hardware
//
#include <hip/hip_runtime.h>

typedef __attribute__((ext_vector_type(16))) _Float16 v16h;
typedef __attribute__((ext_vector_type(8)))  _Float16 v8h;
typedef __attribute__((ext_vector_type(4)))  _Float16 v4h;
typedef __attribute__((ext_vector_type(8)))  float    v8f;

#define M_DIM 128
#define K_DIM 4096
#define N_DIM 16384
#define KC    256              // K chunk staged per block
#define BN    64               // N width per block (8 waves share it)
#define NW    4                // 16-wide n-tiles per wave
#define SW_STRIDE (KC + 8)     // 264 halfs: row step = 4 banks -> 16 rows conflict-free
#define BUFH (BN * SW_STRIDE)  // one LDS buffer in halfs

union V16 { v16h v; v8h h[2]; };

// two packed dwords of int8 -> 8 exact f16 values
__device__ __forceinline__ v8h cvt8(int a, int b) {
  v8h r;
  r[0] = (_Float16)(signed char)(a);
  r[1] = (_Float16)(signed char)(a >> 8);
  r[2] = (_Float16)(signed char)(a >> 16);
  r[3] = (_Float16)(signed char)(a >> 24);
  r[4] = (_Float16)(signed char)(b);
  r[5] = (_Float16)(signed char)(b >> 8);
  r[6] = (_Float16)(signed char)(b >> 16);
  r[7] = (_Float16)(signed char)(b >> 24);
  return r;
}

// x fp32 [128*4096] -> f16 in workspace (read 2 MB, write 1 MB, once)
__global__ __launch_bounds__(256) void l8_cvt_x(const float* __restrict__ x,
                                                _Float16* __restrict__ xh) {
  int i = (blockIdx.x * 256 + threadIdx.x) * 4;
  float4 f = *(const float4*)(x + i);
  v4h h;
  h[0] = (_Float16)f.x; h[1] = (_Float16)f.y;
  h[2] = (_Float16)f.z; h[3] = (_Float16)f.w;
  *(v4h*)(xh + i) = h;
}

__global__ __launch_bounds__(256) void l8_gemm(
    const _Float16* __restrict__ xh,    // [128, 4096] f16
    const signed char* __restrict__ w,  // [16384, 4096] int8, row-major
    const float* __restrict__ scales,   // [16384]
    const float* __restrict__ bias,     // [16384]
    float* __restrict__ out)            // [128, 16384] f32
{
  __shared__ _Float16 sW[2 * BUFH];     // double-buffered dequantized W tile

  const int tid  = threadIdx.x;
  const int lane = tid & 31;
  const int wave = tid >> 5;            // 0..7 -> m-tile
  const int col  = lane & 15;
  const int hi   = lane >> 4;           // lane-group select
  const int n0   = blockIdx.x * BN;
  const int m0   = wave * 16;

  v8f acc[NW];
  const v8f vzero = {0.f, 0.f, 0.f, 0.f, 0.f, 0.f, 0.f, 0.f};
#pragma unroll
  for (int t = 0; t < NW; ++t) acc[t] = vzero;

  // staging: 4 threads per W row, 64 contiguous bytes each (row = 256 B of K)
  const int srow = tid >> 2;            // 0..63
  const int sq   = tid & 3;             // 64-byte quarter
  const signed char* wrow = w + (size_t)(n0 + srow) * K_DIM + sq * 64;
  const _Float16*    arow = xh + (size_t)(m0 + col) * K_DIM;

  // prologue: load chunk 0 into registers
  int4 r0 = *(const int4*)(wrow + 0);
  int4 r1 = *(const int4*)(wrow + 16);
  int4 r2 = *(const int4*)(wrow + 32);
  int4 r3 = *(const int4*)(wrow + 48);

  int p = 0;
  for (int k0 = 0; k0 < K_DIM; k0 += KC) {
    // ---- dequant regs (chunk k0) -> LDS buffer p ----
    _Float16* d = &sW[p * BUFH + srow * SW_STRIDE + sq * 64];
    *(v8h*)(d + 0)  = cvt8(r0.x, r0.y);
    *(v8h*)(d + 8)  = cvt8(r0.z, r0.w);
    *(v8h*)(d + 16) = cvt8(r1.x, r1.y);
    *(v8h*)(d + 24) = cvt8(r1.z, r1.w);
    *(v8h*)(d + 32) = cvt8(r2.x, r2.y);
    *(v8h*)(d + 40) = cvt8(r2.z, r2.w);
    *(v8h*)(d + 48) = cvt8(r3.x, r3.y);
    *(v8h*)(d + 56) = cvt8(r3.z, r3.w);

    // ---- issue next chunk's global loads (wrapped on last iter: uniform pipe) ----
    int k1 = (k0 + KC) & (K_DIM - 1);
    r0 = *(const int4*)(wrow + k1);
    r1 = *(const int4*)(wrow + k1 + 16);
    r2 = *(const int4*)(wrow + k1 + 32);
    r3 = *(const int4*)(wrow + k1 + 48);
    __builtin_prefetch(wrow + ((k0 + 2 * KC) & (K_DIM - 1)), 0, 1);

    __syncthreads();                     // buffer p visible to all waves

    // ---- compute chunk k0 from buffer p (next chunk's loads in flight) ----
    const _Float16* sbuf = &sW[p * BUFH];
#pragma unroll
    for (int kk = 0; kk < KC; kk += 32) {
      // A 16x32 f16 fragment: lane<16 -> K {0..7,16..23}, lane>=16 -> {8..15,24..31}
      V16 a;
      const _Float16* ap = arow + k0 + kk + hi * 8;
      a.h[0] = *(const v8h*)(ap);
      a.h[1] = *(const v8h*)(ap + 16);
      // load ALL B fragments first so ds requests pipeline ahead of the WMMAs
      V16 b[NW];
#pragma unroll
      for (int t = 0; t < NW; ++t) {
        const _Float16* bp = sbuf + (t * 16 + col) * SW_STRIDE + kk + hi * 8;
        b[t].h[0] = *(const v8h*)(bp);
        b[t].h[1] = *(const v8h*)(bp + 16);
      }
#pragma unroll
      for (int t = 0; t < NW; ++t) {
        acc[t] = __builtin_amdgcn_wmma_f32_16x16x32_f16(
            false, a.v, false, b[t].v, (short)0, acc[t], false, false);
      }
    }
    p ^= 1;
  }

  // ---- epilogue: D = acc * (scale/127) + bias, f32 store ----
#pragma unroll
  for (int t = 0; t < NW; ++t) {
    int n = n0 + t * 16 + col;
    float s  = scales[n] * (1.0f / 127.0f);
    float bb = bias[n];
#pragma unroll
    for (int i = 0; i < 8; ++i) {
      int m = m0 + i + hi * 8;   // C/D layout: VGPR i -> M = i (+8 for lanes 16..31)
      out[(size_t)m * N_DIM + n] = acc[t][i] * s + bb;
    }
  }
}

extern "C" void kernel_launch(void* const* d_in, const int* in_sizes, int n_in,
                              void* d_out, int out_size, void* d_ws, size_t ws_size,
                              hipStream_t stream) {
  const float*       x      = (const float*)d_in[0];
  const signed char* w      = (const signed char*)d_in[1];
  const float*       scales = (const float*)d_in[2];
  const float*       bias   = (const float*)d_in[3];
  float*             out    = (float*)d_out;
  _Float16*          xh     = (_Float16*)d_ws;   // needs 128*4096*2 = 1 MiB scratch

  (void)in_sizes; (void)n_in; (void)out_size; (void)ws_size;

  // x fp32 -> f16 once (524288 elems, 4 per thread)
  l8_cvt_x<<<(M_DIM * K_DIM / 4) / 256, 256, 0, stream>>>(x, xh);
  // 256 blocks x 256 threads; each block owns a disjoint 64-wide N strip
  l8_gemm<<<N_DIM / BN, 256, 0, stream>>>(xh, w, scales, bias, out);
}
